// CuboidSelfAttentionLayer_3143916061403
// MI455X (gfx1250) — matrix operand order repaired and checked by reading, not compiled, not run on hardware
//
#include <hip/hip_runtime.h>
#include <hip/hip_bf16.h>
#include <stdint.h>

typedef __attribute__((ext_vector_type(16))) __bf16 v16bf;
typedef __attribute__((ext_vector_type(8)))  float  v8f;
typedef __attribute__((ext_vector_type(8)))  __bf16 v8bf;
typedef unsigned int v4u  __attribute__((ext_vector_type(4)));
typedef int          v8i_ __attribute__((ext_vector_type(8)));
typedef int          v4i_ __attribute__((ext_vector_type(4)));

#define B_   4
#define T_   16
#define H_   56
#define W_   56
#define C_   256
#define NH_  8
#define HD_  32
#define NC_  512
#define VOL  98
#define VP   112
#define SCALE 0.17677669529663687f

// LDS strides (elements) -- padded for conflict-free 16B row chunks
#define SX_STR  264   // X / O tile: 112 x 256 (264 = 256 + TDM pad of 4 dwords)
#define SQ_STR  40    // Q, K, Oh:   112 x 32
#define SVT_STR 136   // Vt:          32 x 128 (cols 98..127 zero)
#define SP_STR  136   // P:          112 x 128
#define SS_STR  100   // S (f32):    112 x 98

union U16 { v16bf v; uint4 q[2]; };
union PK2 { __bf16 h[2]; unsigned int u; };

__device__ __forceinline__ v8f wmma_bf16(v16bf a, v16bf b, v8f c) {
  return __builtin_amdgcn_wmma_f32_16x16x32_bf16(false, a, false, b,
                                                 (short)0, c, false, false);
}

// TDM: DMA a 98x256 bf16 row-major tile into LDS with 4-dword padding per
// 128-dword row => LDS row stride 264 bf16. Descriptor per ISA 08 §8.3/8.4.
__device__ __forceinline__ void tdm_load_98x256_bf16(__bf16* lds_dst,
                                                     const __bf16* gsrc) {
  unsigned lds_addr = (unsigned)(uintptr_t)lds_dst;            // addr[31:0] = LDS offset
  unsigned long long ga = (unsigned long long)(uintptr_t)gsrc;
  v4u g0;
  g0[0] = 1u;                                                  // count=1, user mode
  g0[1] = lds_addr;                                            // lds_addr [63:32]
  g0[2] = (unsigned)(ga & 0xFFFFFFFFull);                      // global_addr lo
  g0[3] = (unsigned)((ga >> 32) & 0x01FFFFFFull) | 0x80000000u; // addr hi | type=2
  v8i_ g1;
  g1[0] = (int)((1u << 16) |   // data_size = 1 -> 2 bytes
                (1u << 20) |   // pad_enable
                (6u << 22) |   // pad_interval code 6 -> every 128 dwords
                (3u << 25));   // pad_amount  code 3 -> 4 dwords
  g1[1] = (int)(256u << 16);   // tensor_dim0 = 256 (bits 79:48, lo16)
  g1[2] = (int)(98u  << 16);   // tensor_dim1 = 98  (bits 111:80, lo16)
  g1[3] = (int)(256u << 16);   // tile_dim0   = 256 (bits 127:112)
  g1[4] = (int)98u;            // tile_dim1   = 98  (bits 143:128)
  g1[5] = (int)256u;           // tensor_dim0_stride = 256 (bits 207:160 lo32)
  g1[6] = 0; g1[7] = 0;
  v4i_ z4 = {0, 0, 0, 0};
  v8i_ z8 = {0, 0, 0, 0, 0, 0, 0, 0};
  __builtin_amdgcn_tensor_load_to_lds(g0, g1, z4, z4, z8, 0);
}

// ---------------- weight conversion fp32 -> bf16 ----------------
__global__ __launch_bounds__(256) void conv_w_kernel(
    const float* __restrict__ wq, const float* __restrict__ wp,
    __bf16* __restrict__ wqb, __bf16* __restrict__ wpb) {
  int i = blockIdx.x * 256 + threadIdx.x;
  if (i < 3 * C_ * C_) wqb[i] = (__bf16)wq[i];
  if (i < C_ * C_)     wpb[i] = (__bf16)wp[i];
}

// ------- LayerNorm + roll(-1,-3,-3) + cuboid reorder -> bf16 -------
__global__ __launch_bounds__(256) void prep_kernel(
    const float* __restrict__ x, const float* __restrict__ gamma,
    const float* __restrict__ beta, __bf16* __restrict__ rA) {
  const int tid  = threadIdx.x;
  const int lane = tid & 31;
  const int r    = blockIdx.x * 8 + (tid >> 5);     // row in (B*nc*vol)

  int b   = r / (NC_ * VOL);
  int rem = r % (NC_ * VOL);
  int cu  = rem / VOL;
  int j   = rem % VOL;
  int ct = cu >> 6, ch = (cu >> 3) & 7, cw = cu & 7;
  int it = j / 49, r2 = j % 49, ih = r2 / 7, iw = r2 % 7;
  int tt = (ct * 2 + it + 1) & 15;
  int hh = ch * 7 + ih + 3; if (hh >= H_) hh -= H_;
  int ww = cw * 7 + iw + 3; if (ww >= W_) ww -= W_;

  const float* src = x + ((((size_t)b * T_ + tt) * H_ + hh) * W_ + ww) * C_;
  int c0 = lane * 8;
  float4 v0 = *(const float4*)(src + c0);
  float4 v1 = *(const float4*)(src + c0 + 4);
  float s  = v0.x + v0.y + v0.z + v0.w + v1.x + v1.y + v1.z + v1.w;
  float ss = v0.x*v0.x + v0.y*v0.y + v0.z*v0.z + v0.w*v0.w +
             v1.x*v1.x + v1.y*v1.y + v1.z*v1.z + v1.w*v1.w;
  for (int o = 16; o > 0; o >>= 1) {
    s  += __shfl_xor(s,  o, 32);
    ss += __shfl_xor(ss, o, 32);
  }
  float mean = s * (1.0f / 256.0f);
  float var  = ss * (1.0f / 256.0f) - mean * mean;
  float rstd = rsqrtf(var + 1e-5f);

  float4 g0 = *(const float4*)(gamma + c0);
  float4 g1 = *(const float4*)(gamma + c0 + 4);
  float4 e0 = *(const float4*)(beta + c0);
  float4 e1 = *(const float4*)(beta + c0 + 4);

  v8bf o;
  o[0] = (__bf16)((v0.x - mean) * rstd * g0.x + e0.x);
  o[1] = (__bf16)((v0.y - mean) * rstd * g0.y + e0.y);
  o[2] = (__bf16)((v0.z - mean) * rstd * g0.z + e0.z);
  o[3] = (__bf16)((v0.w - mean) * rstd * g0.w + e0.w);
  o[4] = (__bf16)((v1.x - mean) * rstd * g1.x + e1.x);
  o[5] = (__bf16)((v1.y - mean) * rstd * g1.y + e1.y);
  o[6] = (__bf16)((v1.z - mean) * rstd * g1.z + e1.z);
  o[7] = (__bf16)((v1.w - mean) * rstd * g1.w + e1.w);
  *(v8bf*)(rA + (size_t)r * C_ + c0) = o;
}

// ------------- fused QKV + masked attention per cuboid -------------
__global__ __launch_bounds__(256) void attn_kernel(
    const __bf16* __restrict__ rA, const __bf16* __restrict__ wqkv,
    __bf16* __restrict__ Og) {
  __shared__ __align__(16) __bf16 sX [VP  * SX_STR];
  __shared__ __align__(16) __bf16 sQ [VP  * SQ_STR];
  __shared__ __align__(16) __bf16 sK [VP  * SQ_STR];
  __shared__ __align__(16) __bf16 sVt[HD_ * SVT_STR];
  __shared__ __align__(16) __bf16 sP [VP  * SP_STR];
  __shared__ __align__(16) float  sS [VP  * SS_STR];   // reused as Oh (bf16)
  __shared__ int sRid[VP];
  __bf16* sOh = (__bf16*)sS;

  const int tid  = threadIdx.x;
  const int wg   = blockIdx.x;       // b*512 + cuboid
  const int lane = tid & 31;
  const int wave = tid >> 5;
  const int kh   = lane >> 4;
  const int ln   = lane & 15;
  const int cu   = wg & 511;
  const int ct = cu >> 6, ch = (cu >> 3) & 7, cw = cu & 7;

  // ---- async TDM load of X tile (98x256) into LDS rows 0..97 ----
  if (wave == 0)
    tdm_load_98x256_bf16(sX, rA + (size_t)wg * (VOL * C_));
  // zero pad rows 98..111 of sX
  for (int slot = tid; slot < (VP - VOL) * 16; slot += 256) {
    int row = VOL + (slot >> 4), ck = slot & 15;
    uint4* dst = (uint4*)(&sX[row * SX_STR + ck * 16]);
    uint4 zz = make_uint4(0u, 0u, 0u, 0u);
    dst[0] = zz; dst[1] = zz;
  }
  // zero Vt (keeps padded columns 98..135 at zero)
  for (int i = tid; i < (HD_ * SVT_STR) / 2; i += 256)
    ((unsigned int*)sVt)[i] = 0u;
  // shift-window region ids
  if (tid < VP) {
    int rid = -1;
    if (tid < VOL) {
      int it = tid / 49, r2 = tid % 49, ih = r2 / 7, iw = r2 % 7;
      int ts = ct * 2 + it, hs = ch * 7 + ih, ws2 = cw * 7 + iw;
      int rt = ts < 14 ? 0 : (ts < 15 ? 1 : 2);
      int rh = hs < 49 ? 0 : (hs < 53 ? 1 : 2);
      int rw = ws2 < 49 ? 0 : (ws2 < 53 ? 1 : 2);
      rid = rt * 9 + rh * 3 + rw;
    }
    sRid[tid] = rid;
  }
  if (wave == 0) __builtin_amdgcn_s_wait_tensorcnt(0);
  __syncthreads();

  for (int h = 0; h < NH_; ++h) {
    // ---- phase B: QKV slices, three uniform passes (no divergent stores) --
    // Q tiles
    for (int t = wave; t < 14; t += 8) {
      int mt = t >> 1, nt2 = t & 1;
      int within = nt2 * 16 + ln;
      const __bf16* Arow = &sX[(mt * 16 + ln) * SX_STR];
      const __bf16* Brow = wqkv + (size_t)(h * HD_ + within) * C_;
      __builtin_prefetch((const void*)Brow, 0, 0);
      v8f acc = {};
#pragma unroll
      for (int ks = 0; ks < 8; ++ks) {
        U16 a, bb;
        a.q[0]  = *(const uint4*)(Arow + ks * 32 + kh * 8);
        a.q[1]  = *(const uint4*)(Arow + ks * 32 + kh * 8 + 16);
        bb.q[0] = *(const uint4*)(Brow + ks * 32 + kh * 8);
        bb.q[1] = *(const uint4*)(Brow + ks * 32 + kh * 8 + 16);
        acc = wmma_bf16(a.v, bb.v, acc);
      }
#pragma unroll
      for (int g = 0; g < 8; ++g)
        sQ[(mt * 16 + g + 8 * kh) * SQ_STR + within] = (__bf16)(acc[g] * SCALE);
    }
    // K tiles
    for (int t = wave; t < 14; t += 8) {
      int mt = t >> 1, nt2 = t & 1;
      int within = nt2 * 16 + ln;
      const __bf16* Arow = &sX[(mt * 16 + ln) * SX_STR];
      const __bf16* Brow = wqkv + (size_t)(C_ + h * HD_ + within) * C_;
      __builtin_prefetch((const void*)Brow, 0, 0);
      v8f acc = {};
#pragma unroll
      for (int ks = 0; ks < 8; ++ks) {
        U16 a, bb;
        a.q[0]  = *(const uint4*)(Arow + ks * 32 + kh * 8);
        a.q[1]  = *(const uint4*)(Arow + ks * 32 + kh * 8 + 16);
        bb.q[0] = *(const uint4*)(Brow + ks * 32 + kh * 8);
        bb.q[1] = *(const uint4*)(Brow + ks * 32 + kh * 8 + 16);
        acc = wmma_bf16(a.v, bb.v, acc);
      }
#pragma unroll
      for (int g = 0; g < 8; ++g)
        sK[(mt * 16 + g + 8 * kh) * SQ_STR + within] = (__bf16)acc[g];
    }
    // V tiles -> transposed store, packed pairs (adjacent m => one b32 store)
    for (int t = wave; t < 14; t += 8) {
      int mt = t >> 1, nt2 = t & 1;
      int within = nt2 * 16 + ln;
      const __bf16* Arow = &sX[(mt * 16 + ln) * SX_STR];
      const __bf16* Brow = wqkv + (size_t)(2 * C_ + h * HD_ + within) * C_;
      __builtin_prefetch((const void*)Brow, 0, 0);
      v8f acc = {};
#pragma unroll
      for (int ks = 0; ks < 8; ++ks) {
        U16 a, bb;
        a.q[0]  = *(const uint4*)(Arow + ks * 32 + kh * 8);
        a.q[1]  = *(const uint4*)(Arow + ks * 32 + kh * 8 + 16);
        bb.q[0] = *(const uint4*)(Brow + ks * 32 + kh * 8);
        bb.q[1] = *(const uint4*)(Brow + ks * 32 + kh * 8 + 16);
        acc = wmma_bf16(a.v, bb.v, acc);
      }
#pragma unroll
      for (int g = 0; g < 8; g += 2) {
        int m = mt * 16 + g + 8 * kh;           // even
        PK2 p;
        p.h[0] = (__bf16)acc[g];
        p.h[1] = (__bf16)acc[g + 1];
        *(unsigned int*)&sVt[within * SVT_STR + m] = p.u;
      }
    }
    __syncthreads();

    // ---- phase C: S = Q*K^T (M=112, N=112, K=32) ----
    for (int t = wave; t < 49; t += 8) {
      int mt = t / 7, nt = t % 7;
      const __bf16* Arow = &sQ[(mt * 16 + ln) * SQ_STR];
      const __bf16* Brow = &sK[(nt * 16 + ln) * SQ_STR];
      U16 a, bb;
      a.q[0]  = *(const uint4*)(Arow + kh * 8);
      a.q[1]  = *(const uint4*)(Arow + kh * 8 + 16);
      bb.q[0] = *(const uint4*)(Brow + kh * 8);
      bb.q[1] = *(const uint4*)(Brow + kh * 8 + 16);
      v8f acc = {};
      acc = wmma_bf16(a.v, bb.v, acc);
      int n = nt * 16 + ln;
      if (n < VOL) {
#pragma unroll
        for (int g = 0; g < 8; ++g)
          sS[(mt * 16 + g + 8 * kh) * SS_STR + n] = acc[g];
      }
    }
    __syncthreads();

    // ---- phase D: masked softmax -> P (bf16, 128 padded cols) ----
    if (tid < VP) {
      int i = tid;
      if (i < VOL) {
        int myrid = sRid[i];
        const float* srow = &sS[i * SS_STR];
        float mx = -3.0e38f;
        for (int j = 0; j < VOL; ++j)
          if (sRid[j] == myrid) mx = fmaxf(mx, srow[j]);
        float sum = 0.0f;
        for (int j = 0; j < VOL; ++j)
          if (sRid[j] == myrid) sum += __expf(srow[j] - mx);
        float rinv = 1.0f / sum;
        for (int j = 0; j < 128; ++j) {
          float p = 0.0f;
          if (j < VOL && sRid[j] == myrid) p = __expf(srow[j] - mx) * rinv;
          sP[i * SP_STR + j] = (__bf16)p;
        }
      } else {
        for (int j = 0; j < 128; ++j) sP[i * SP_STR + j] = (__bf16)0.0f;
      }
    }
    __syncthreads();

    // ---- phase E: O_h = P*V (M=112, N=32, K=128) ----
    for (int t = wave; t < 14; t += 8) {
      int mt = t >> 1, nt = t & 1;
      const __bf16* Arow = &sP[(mt * 16 + ln) * SP_STR];
      const __bf16* Brow = &sVt[(nt * 16 + ln) * SVT_STR];
      v8f acc = {};
#pragma unroll
      for (int ks = 0; ks < 4; ++ks) {
        U16 a, bb;
        a.q[0]  = *(const uint4*)(Arow + ks * 32 + kh * 8);
        a.q[1]  = *(const uint4*)(Arow + ks * 32 + kh * 8 + 16);
        bb.q[0] = *(const uint4*)(Brow + ks * 32 + kh * 8);
        bb.q[1] = *(const uint4*)(Brow + ks * 32 + kh * 8 + 16);
        acc = wmma_bf16(a.v, bb.v, acc);
      }
      int n = nt * 16 + ln;
#pragma unroll
      for (int g = 0; g < 8; ++g)
        sOh[(mt * 16 + g + 8 * kh) * SQ_STR + n] = (__bf16)acc[g];
    }
    __syncthreads();

    // ---- store O_h to global (bf16) ----
    for (int slot = tid; slot < VOL * 2; slot += 256) {
      int v = slot >> 1, hf = slot & 1;
      const uint4* src = (const uint4*)(&sOh[v * SQ_STR + hf * 16]);
      uint4* dst = (uint4*)(Og + ((size_t)wg * VOL + v) * C_ + h * HD_ + hf * 16);
      dst[0] = src[0]; dst[1] = src[1];
    }
    __syncthreads();
  }
}

// -------- final projection + reverse reorder + roll-back --------
__global__ __launch_bounds__(256) void proj_kernel(
    const __bf16* __restrict__ Og, const __bf16* __restrict__ wproj,
    const float* __restrict__ bproj, float* __restrict__ out) {
  __shared__ __align__(16) __bf16 sO[VP * SX_STR];

  const int tid  = threadIdx.x;
  const int wg   = blockIdx.x;
  const int lane = tid & 31;
  const int wave = tid >> 5;
  const int kh   = lane >> 4;
  const int ln   = lane & 15;
  const int b  = wg >> 9, cu = wg & 511;
  const int ct = cu >> 6, ch = (cu >> 3) & 7, cw = cu & 7;

  if (wave == 0)
    tdm_load_98x256_bf16(sO, Og + (size_t)wg * (VOL * C_));
  for (int slot = tid; slot < (VP - VOL) * 16; slot += 256) {
    int row = VOL + (slot >> 4), ck = slot & 15;
    uint4* dst = (uint4*)(&sO[row * SX_STR + ck * 16]);
    uint4 zz = make_uint4(0u, 0u, 0u, 0u);
    dst[0] = zz; dst[1] = zz;
  }
  if (wave == 0) __builtin_amdgcn_s_wait_tensorcnt(0);
  __syncthreads();

  for (int t = wave; t < 112; t += 8) {
    int mt = t >> 4, nt = t & 15;
    const __bf16* Arow = &sO[(mt * 16 + ln) * SX_STR];
    const __bf16* Brow = wproj + (size_t)(nt * 16 + ln) * C_;
    __builtin_prefetch((const void*)Brow, 0, 0);
    v8f acc = {};
#pragma unroll
    for (int ks = 0; ks < 8; ++ks) {
      U16 a, bb;
      a.q[0]  = *(const uint4*)(Arow + ks * 32 + kh * 8);
      a.q[1]  = *(const uint4*)(Arow + ks * 32 + kh * 8 + 16);
      bb.q[0] = *(const uint4*)(Brow + ks * 32 + kh * 8);
      bb.q[1] = *(const uint4*)(Brow + ks * 32 + kh * 8 + 16);
      acc = wmma_bf16(a.v, bb.v, acc);
    }
    int n = nt * 16 + ln;
    float bias = bproj[n];
#pragma unroll
    for (int g = 0; g < 8; ++g) {
      int m = mt * 16 + g + 8 * kh;
      if (m < VOL) {
        int it = m / 49, r2 = m % 49, ih = r2 / 7, iw = r2 % 7;
        int tt = (ct * 2 + it + 1) & 15;
        int hh = ch * 7 + ih + 3; if (hh >= H_) hh -= H_;
        int ww = cw * 7 + iw + 3; if (ww >= W_) ww -= W_;
        size_t idx = ((((size_t)b * T_ + tt) * H_ + hh) * W_ + ww) * C_ + n;
        out[idx] = acc[g] + bias;
      }
    }
  }
}

extern "C" void kernel_launch(void* const* d_in, const int* in_sizes, int n_in,
                              void* d_out, int out_size, void* d_ws, size_t ws_size,
                              hipStream_t stream) {
  const float* x     = (const float*)d_in[0];
  const float* gamma = (const float*)d_in[1];
  const float* beta  = (const float*)d_in[2];
  const float* wqkv  = (const float*)d_in[3];
  const float* wproj = (const float*)d_in[4];
  const float* bproj = (const float*)d_in[5];
  float* out = (float*)d_out;

  char* ws = (char*)d_ws;
  const size_t RA_BYTES = (size_t)B_ * NC_ * VOL * C_ * sizeof(__bf16); // 102,760,448
  __bf16* rA  = (__bf16*)(ws);
  __bf16* Og  = (__bf16*)(ws + RA_BYTES);
  __bf16* wqb = (__bf16*)(ws + 2 * RA_BYTES);
  __bf16* wpb = (__bf16*)(ws + 2 * RA_BYTES + (size_t)3 * C_ * C_ * sizeof(__bf16));

  conv_w_kernel<<<768, 256, 0, stream>>>(wqkv, wproj, wqb, wpb);
  prep_kernel<<<(B_ * NC_ * VOL) / 8, 256, 0, stream>>>(x, gamma, beta, rA);
  attn_kernel<<<B_ * NC_, 256, 0, stream>>>(rA, wqb, Og);
  proj_kernel<<<B_ * NC_, 256, 0, stream>>>(Og, wpb, bproj, out);
}
